// ScaledDotProductAttention_6493990552224
// MI455X (gfx1250) — matrix operand order repaired and checked by reading, not compiled, not run on hardware
//
#include <hip/hip_runtime.h>

typedef __attribute__((ext_vector_type(16))) _Float16 v16h;
typedef __attribute__((ext_vector_type(8)))  float    v8f;

#define B_    32
#define S_    2048
#define DK_   64
#define DV_   64
#define SCP   2052          // padded LDS row stride (floats): 2052 % 64 = 4 -> no bank conflicts
#define WAVES 8
#define KPW   (S_ / WAVES)  // 256 keys per wave

// One workgroup = one (batch, 16-query-row) tile. 8 waves split the 2048-key axis.
__global__ __launch_bounds__(256)
void sdpa_wmma_kernel(const float* __restrict__ qg, const float* __restrict__ kg_,
                      const float* __restrict__ vg, const unsigned char* __restrict__ mg,
                      float* __restrict__ outg, float* __restrict__ attng)
{
    extern __shared__ float lds[];
    float* sc     = lds;              // [16][SCP]  exp(scores), later normalized P
    float* rowsum = lds + 16 * SCP;   // [16]
    float* rinv   = rowsum + 16;      // [16]
    float* ored   = rinv + 16;        // [WAVES][16][64]  P*V partial tiles

    const int tid   = threadIdx.x;
    const int lane  = tid & 31;
    const int wave  = tid >> 5;
    const int b     = blockIdx.x >> 7;     // batch
    const int qt    = blockIdx.x & 127;    // query tile
    const int qrow0 = qt << 4;

    if (tid < 16) rowsum[tid] = 0.f;
    __syncthreads();

    // WMMA fragment coordinates (per ISA 7.12.2, wave32)
    const int mrow = lane & 15;            // A: row M
    const int kgrp = (lane >> 4) << 3;     // A: K sub-group (0 or 8)
    const int ncol = lane & 15;            // C/D: column N
    const int mtop = (lane >> 4) << 3;     // C/D: M base (0 or 8)

    // ---- Load Q tile as two 16x32 f16 A-fragments (depth 0-31, 32-63) ----
    const float* qbase = qg + ((size_t)b * S_ + qrow0) * DK_;
    v16h a0, a1;
    #pragma unroll
    for (int h = 0; h < 8; ++h) {
        a0[h]     = (_Float16)qbase[mrow * DK_ + kgrp + h];
        a0[h + 8] = (_Float16)qbase[mrow * DK_ + 16 + kgrp + h];
        a1[h]     = (_Float16)qbase[mrow * DK_ + 32 + kgrp + h];
        a1[h + 8] = (_Float16)qbase[mrow * DK_ + 48 + kgrp + h];
    }

    const float* kbase = kg_ + (size_t)b * S_ * DK_;
    const unsigned char* mbase = mg + ((size_t)b * S_ + qrow0) * S_;

    float rpart[8];
    #pragma unroll
    for (int r = 0; r < 8; ++r) rpart[r] = 0.f;

    // ---- Pass 1: S = exp(mask(QK^T / 8)) into LDS, accumulate row sums ----
    const int key_base = wave * KPW;
    for (int j = 0; j < KPW / 16; ++j) {
        const int key0 = key_base + (j << 4);
        // B = K^T (32 x 16): lane%16 = key column, elements = depth in table order (kgrp sub-group).
        v16h b0, b1;
        #pragma unroll
        for (int h = 0; h < 8; ++h) {
            b0[h]     = (_Float16)kbase[(size_t)(key0 + ncol) * DK_ + kgrp + h];
            b0[h + 8] = (_Float16)kbase[(size_t)(key0 + ncol) * DK_ + 16 + kgrp + h];
            b1[h]     = (_Float16)kbase[(size_t)(key0 + ncol) * DK_ + 32 + kgrp + h];
            b1[h + 8] = (_Float16)kbase[(size_t)(key0 + ncol) * DK_ + 48 + kgrp + h];
        }
        v8f c = {};
        c = __builtin_amdgcn_wmma_f32_16x16x32_f16(false, a0, false, b0, (short)0, c, false, false);
        c = __builtin_amdgcn_wmma_f32_16x16x32_f16(false, a1, false, b1, (short)0, c, false, false);
        #pragma unroll
        for (int r = 0; r < 8; ++r) {
            const int m = mtop + r;
            const float s = c[r] * 0.125f;                       // 1/TEMPERATURE
            const float e = mbase[(size_t)m * S_ + key0 + ncol] ? 0.f : __expf(s);
            sc[m * SCP + key0 + ncol] = e;
            rpart[r] += e;
        }
    }
    #pragma unroll
    for (int r = 0; r < 8; ++r) atomicAdd(&rowsum[mtop + r], rpart[r]);  // ds_add_f32
    __syncthreads();
    if (tid < 16) rinv[tid] = 1.f / rowsum[tid];
    __syncthreads();

    // ---- Normalize in LDS and stream attn to HBM (float4, fully coalesced) ----
    float* attn_row = attng + ((size_t)b * S_ + qrow0) * S_;
    for (int m = 0; m < 16; ++m) {
        const float inv = rinv[m];
        #pragma unroll
        for (int c0 = 0; c0 < S_; c0 += 1024) {
            const int col = c0 + (tid << 2);
            float4 p = *(float4*)&sc[m * SCP + col];
            p.x *= inv; p.y *= inv; p.z *= inv; p.w *= inv;
            *(float4*)&sc[m * SCP + col] = p;
            *(float4*)&attn_row[(size_t)m * S_ + col] = p;
        }
    }
    __syncthreads();

    // ---- Pass 2: O = P * V, each wave over its 256 keys, 4 N-tiles of DV ----
    const float* vbase = vg + (size_t)b * S_ * DV_;
    v8f acc[4] = {};
    for (int s0 = 0; s0 < KPW; s0 += 32) {
        const int k0 = key_base + s0;
        v16h ap;                           // A = P tile 16x32 from LDS
        #pragma unroll
        for (int h = 0; h < 8; ++h) {
            ap[h]     = (_Float16)sc[mrow * SCP + k0 + kgrp + h];
            ap[h + 8] = (_Float16)sc[mrow * SCP + k0 + 16 + kgrp + h];
        }
        const float* vrow = vbase + (size_t)k0 * DV_;   // B: lane%16 = V column (N), elements = V rows (K) in table order
        v16h bv[4];
        #pragma unroll
        for (int t = 0; t < 4; ++t)
            #pragma unroll
            for (int h = 0; h < 16; ++h)
                bv[t][h] = (_Float16)vrow[(size_t)(kgrp + (h & 7) + ((h >> 3) << 4)) * DV_ + t * 16 + ncol];
        #pragma unroll
        for (int t = 0; t < 4; ++t)
            acc[t] = __builtin_amdgcn_wmma_f32_16x16x32_f16(false, ap, false, bv[t],
                                                            (short)0, acc[t], false, false);
    }

    // ---- Cross-wave reduction of the 16x64 output tile through LDS ----
    #pragma unroll
    for (int t = 0; t < 4; ++t)
        #pragma unroll
        for (int r = 0; r < 8; ++r)
            ored[wave * 1024 + (mtop + r) * 64 + t * 16 + ncol] = acc[t][r];
    __syncthreads();

    float* obase = outg + ((size_t)b * S_ + qrow0) * DV_;
    for (int idx = tid; idx < 1024; idx += 256) {   // idx = m*64 + n, contiguous in out
        float sum = 0.f;
        #pragma unroll
        for (int w = 0; w < WAVES; ++w) sum += ored[w * 1024 + idx];
        obase[idx] = sum;
    }
}

extern "C" void kernel_launch(void* const* d_in, const int* in_sizes, int n_in,
                              void* d_out, int out_size, void* d_ws, size_t ws_size,
                              hipStream_t stream) {
    (void)in_sizes; (void)n_in; (void)out_size; (void)d_ws; (void)ws_size;
    const float*         q    = (const float*)d_in[0];
    const float*         k    = (const float*)d_in[1];
    const float*         v    = (const float*)d_in[2];
    const unsigned char* mask = (const unsigned char*)d_in[3];   // jnp.bool_ -> 1 byte

    float* out  = (float*)d_out;                     // [B,S,DV] first
    float* attn = out + (size_t)B_ * S_ * DV_;       // then [B,S,S]

    const size_t lds_bytes = (size_t)(16 * SCP + 16 + 16 + WAVES * 16 * 64) * sizeof(float); // ~160 KB
    hipFuncSetAttribute((const void*)sdpa_wmma_kernel,
                        hipFuncAttributeMaxDynamicSharedMemorySize, (int)lds_bytes);

    sdpa_wmma_kernel<<<dim3(B_ * (S_ / 16)), dim3(256), lds_bytes, stream>>>(
        q, k, v, mask, out, attn);
}
